// RN_10264971837726
// MI455X (gfx1250) — matrix operand order repaired and checked by reading, not compiled, not run on hardware
//
#include <hip/hip_runtime.h>
#include <hip/hip_bf16.h>
#include <stdint.h>

// Problem dims (compile-time constants from the reference)
#define I_CNN 2048
#define QEMB  300
#define QPAD  320
#define HL    1024
#define HID   512
#define BB    256
#define TT    20
#define MM    128
#define LDW   (I_CNN + HL)   // 3072, row stride of W_w / Wp_w

typedef __attribute__((ext_vector_type(16))) __bf16         v16bf;
typedef __attribute__((ext_vector_type(8)))  float          v8f;
typedef __attribute__((ext_vector_type(4)))  unsigned int   v4u;
typedef __attribute__((ext_vector_type(4)))  unsigned short v4us;

// ---------------------------------------------------------------- helpers

__device__ __forceinline__ unsigned short f2bf(float f) {
  // fp32 -> bf16, round-to-nearest-even
  unsigned u = __builtin_bit_cast(unsigned, f);
  u += 0x7FFFu + ((u >> 16) & 1u);
  return (unsigned short)(u >> 16);
}

__device__ __forceinline__ float sigmoidf_(float x) {
  return 1.0f / (1.0f + __expf(-x));
}

__device__ __forceinline__ v8f wmma_bf16(v16bf a, v16bf b, v8f c) {
  // D = A(16x32 bf16) * B(32x16 bf16) + C(16x16 f32)
  return __builtin_amdgcn_wmma_f32_16x16x32_bf16(
      /*neg_a=*/false, a, /*neg_b=*/false, b,
      /*c_mod=*/(short)0, c, /*reuse_a=*/false, /*reuse_b=*/false);
}

struct bfpair { v4u lo; v4u hi; };  // 32 bytes == v16bf

__device__ __forceinline__ v16bf frag_from(const unsigned short* p0,
                                           const unsigned short* p1) {
  bfpair p;
  p.lo = *(const v4u*)p0;   // 16B -> fragment elements 0..7
  p.hi = *(const v4u*)p1;   // 16B -> fragment elements 8..15
  return __builtin_bit_cast(v16bf, p);
}

// A fragment (16x32 bf16, row-major packed source).
// ISA layout: lanes 0-15: M=lane, K={k0+0..7, k0+16..23};
//             lanes 16-31: M=lane-16, K={k0+8..15, k0+24..31}.
__device__ __forceinline__ v16bf load_a_bf(const unsigned short* __restrict__ A,
                                           long lda, int r0, int k0) {
  int lane = threadIdx.x & 31;
  int m = lane & 15, half = lane >> 4;
  const unsigned short* p = A + (long)(r0 + m) * lda + k0 + half * 8;
  return frag_from(p, p + 16);
}

// B fragment: B[k][n] = W[n0+n][k0+k] (row-major packed W, GEMM with W^T).
// lanes 0-15: N=lane, K={k0+0..7, k0+16..23}; lanes 16-31: N=lane-16, K={k0+8..15, k0+24..31}.
__device__ __forceinline__ v16bf load_b_bf(const unsigned short* __restrict__ W,
                                           long ldw, int n0, int k0) {
  int lane = threadIdx.x & 31;
  int n = lane & 15, half = lane >> 4;
  const unsigned short* p = W + (long)(n0 + n) * ldw + k0 + half * 8;
  return frag_from(p, p + 16);
}

// ---------------------------------------------------------------- pack kernels

__global__ void __launch_bounds__(256)
pack_bf16_kernel(const float* __restrict__ in, unsigned short* __restrict__ out,
                 long n) {
  long i = ((long)blockIdx.x * 256 + threadIdx.x) * 4;
  if (i >= n) return;
  float4 v = *(const float4*)(in + i);
  v4us o;
  o[0] = f2bf(v.x); o[1] = f2bf(v.y); o[2] = f2bf(v.z); o[3] = f2bf(v.w);
  *(v4us*)(out + i) = o;
}

// Row-padded fp32 -> bf16 repack: out[r][0..ldout) with zeros for c >= ldin.
__global__ void __launch_bounds__(256)
pack_pad_kernel(const float* __restrict__ in, unsigned short* __restrict__ out,
                int rows, int ldin, int ldout) {
  long i = (long)blockIdx.x * 256 + threadIdx.x;
  long total = (long)rows * ldout;
  if (i >= total) return;
  int r = (int)(i / ldout), c = (int)(i % ldout);
  out[i] = (c < ldin) ? f2bf(in[(long)r * ldin + c]) : (unsigned short)0;
}

// q_feats [B,T,300] fp32 -> xpad [T,B,320] bf16 (t-major, K zero-padded).
__global__ void __launch_bounds__(256)
pack_x_kernel(const float* __restrict__ q, unsigned short* __restrict__ out) {
  long i = (long)blockIdx.x * 256 + threadIdx.x;
  if (i >= (long)TT * BB * QPAD) return;
  int c = (int)(i % QPAD);
  long rb = i / QPAD;
  int b = (int)(rb % BB), t = (int)(rb / BB);
  out[i] = (c < QEMB) ? f2bf(q[((long)b * TT + t) * QEMB + c]) : (unsigned short)0;
}

// ---------------------------------------------------------------- LSTM

struct Frag5 { v16bf a, bi, bf, bg, bo; };

__device__ __forceinline__ Frag5 load5(const unsigned short* __restrict__ A,
                                       long lda, int r0,
                                       const unsigned short* __restrict__ W,
                                       long ldw, int j0, int k0) {
  Frag5 f;
  f.a  = load_a_bf(A, lda, r0, k0);
  f.bi = load_b_bf(W, ldw, 0 * HL + j0, k0);
  f.bf = load_b_bf(W, ldw, 1 * HL + j0, k0);
  f.bg = load_b_bf(W, ldw, 2 * HL + j0, k0);
  f.bo = load_b_bf(W, ldw, 3 * HL + j0, k0);
  return f;
}

#define GATE_WMMA4(F)                    \
  do {                                   \
    ai = wmma_bf16((F).a, (F).bi, ai);   \
    af = wmma_bf16((F).a, (F).bf, af);   \
    ag = wmma_bf16((F).a, (F).bg, ag);   \
    ao = wmma_bf16((F).a, (F).bo, ao);   \
  } while (0)

// One LSTM step: wave = 16(batch) x 16(hidden) tile, four gate accumulators.
// K loop is a 2-deep rotated software pipeline: fragment set A is reloaded
// (two chunks ahead) right after its WMMAs while set B computes — no
// register moves, loads land directly in the WMMA source registers.
__global__ void __launch_bounds__(128)
lstm_step_kernel(const unsigned short* __restrict__ xpad,   // [T,B,320]
                 const unsigned short* __restrict__ w_ihpk, // [4H,320]
                 const unsigned short* __restrict__ w_hhpk, // [4H,1024]
                 const float* __restrict__ b_ih, const float* __restrict__ b_hh,
                 const unsigned short* __restrict__ h_prev, // [B,1024] bf16
                 unsigned short* __restrict__ h_next,       // [B,1024] bf16
                 float* __restrict__ c_buf, int t) {
  int wave = blockIdx.x * (blockDim.x >> 5) + (threadIdx.x >> 5);
  int b0 = (wave >> 6) << 4;  // 64 hidden tiles per batch tile
  int j0 = (wave & 63) << 4;

  v8f ai = {}, af = {}, ag = {}, ao = {};

  const unsigned short* xt = xpad + (long)t * BB * QPAD;

  // x phase: 10 chunks of K=32
  Frag5 fA = load5(xt, QPAD, b0, w_ihpk, QPAD, j0, 0);
  Frag5 fB = load5(xt, QPAD, b0, w_ihpk, QPAD, j0, 32);
#pragma unroll 1
  for (int c = 0; c < 8; c += 2) {
    GATE_WMMA4(fA);
    fA = load5(xt, QPAD, b0, w_ihpk, QPAD, j0, (c + 2) * 32);
    GATE_WMMA4(fB);
    fB = load5(xt, QPAD, b0, w_ihpk, QPAD, j0, (c + 3) * 32);
  }
  // transition: drain last two x chunks while the first h loads fly
  GATE_WMMA4(fA);
  fA = load5(h_prev, HL, b0, w_hhpk, HL, j0, 0);
  GATE_WMMA4(fB);
  fB = load5(h_prev, HL, b0, w_hhpk, HL, j0, 32);
  // h phase: 32 chunks of K=32
#pragma unroll 1
  for (int c = 0; c < 30; c += 2) {
    GATE_WMMA4(fA);
    fA = load5(h_prev, HL, b0, w_hhpk, HL, j0, (c + 2) * 32);
    GATE_WMMA4(fB);
    fB = load5(h_prev, HL, b0, w_hhpk, HL, j0, (c + 3) * 32);
  }
  GATE_WMMA4(fA);
  GATE_WMMA4(fB);

  int lane = threadIdx.x & 31;
  int n = lane & 15, half = lane >> 4;
  float Bi = b_ih[0 * HL + j0 + n] + b_hh[0 * HL + j0 + n];
  float Bf = b_ih[1 * HL + j0 + n] + b_hh[1 * HL + j0 + n];
  float Bg = b_ih[2 * HL + j0 + n] + b_hh[2 * HL + j0 + n];
  float Bo = b_ih[3 * HL + j0 + n] + b_hh[3 * HL + j0 + n];
#pragma unroll
  for (int r = 0; r < 8; ++r) {
    int b = b0 + r + 8 * half;  // C layout: VGPR r -> M=r (lanes<16), r+8 (>=16)
    long idx = (long)b * HL + j0 + n;
    float ig = sigmoidf_(ai[r] + Bi);
    float fg = sigmoidf_(af[r] + Bf);
    float gg = tanhf(ag[r] + Bg);
    float og = sigmoidf_(ao[r] + Bo);
    float c = fg * c_buf[idx] + ig * gg;
    c_buf[idx] = c;
    h_next[idx] = f2bf(og * tanhf(c));
  }
}

// ---------------------------------------------------------------- q projection

struct Frag3 { v16bf a, by, bg; };

__device__ __forceinline__ Frag3 load3(const unsigned short* __restrict__ h,
                                       const unsigned short* __restrict__ Wpk,
                                       const unsigned short* __restrict__ Wppk,
                                       int b0, int n0, int k0) {
  Frag3 f;
  f.a  = load_a_bf(h, HL, b0, k0);
  f.by = load_b_bf(Wpk  + I_CNN, LDW, n0, k0);  // q-part columns
  f.bg = load_b_bf(Wppk + I_CNN, LDW, n0, k0);
  return f;
}

#define QP_WMMA2(F)                        \
  do {                                     \
    ay  = wmma_bf16((F).a, (F).by, ay);    \
    agg = wmma_bf16((F).a, (F).bg, agg);   \
  } while (0)

// qy[b,h] = q_rnn[b] @ W_w[:,2048:]^T + W_b[h]; qg likewise with Wp. K=1024.
__global__ void __launch_bounds__(128)
qproj_kernel(const unsigned short* __restrict__ h,    // [B,1024] bf16
             const unsigned short* __restrict__ Wpk,  // [512,3072] bf16
             const float* __restrict__ W_b,
             const unsigned short* __restrict__ Wppk, const float* __restrict__ Wp_b,
             float* __restrict__ qy, float* __restrict__ qg) {
  int wave = blockIdx.x * (blockDim.x >> 5) + (threadIdx.x >> 5);
  int b0 = (wave >> 5) << 4;  // 32 col tiles (512/16)
  int n0 = (wave & 31) << 4;
  v8f ay = {}, agg = {};

  Frag3 fA = load3(h, Wpk, Wppk, b0, n0, 0);
  Frag3 fB = load3(h, Wpk, Wppk, b0, n0, 32);
#pragma unroll 1
  for (int c = 0; c < 30; c += 2) {
    QP_WMMA2(fA);
    fA = load3(h, Wpk, Wppk, b0, n0, (c + 2) * 32);
    QP_WMMA2(fB);
    fB = load3(h, Wpk, Wppk, b0, n0, (c + 3) * 32);
  }
  QP_WMMA2(fA);
  QP_WMMA2(fB);

  int lane = threadIdx.x & 31;
  int n = lane & 15, half = lane >> 4;
  float by_ = W_b[n0 + n], bg_ = Wp_b[n0 + n];
#pragma unroll
  for (int r = 0; r < 8; ++r) {
    int b = b0 + r + 8 * half;
    qy[(long)b * HID + n0 + n] = ay[r] + by_;
    qg[(long)b * HID + n0 + n] = agg[r] + bg_;
  }
}

// ---------------------------------------------------------------- main GEMM

// rows = B*M = 32768, cols = 512, K = 2048.
// One block per 16-row tile covering ALL 512 columns: box_feats is read from
// HBM exactly once. K=64 staging tiles, double-buffered in LDS (one barrier
// per 64-K), next global stage prefetched under the 16 WMMAs of the current
// tile. 8 waves x 4 col tiles x {Y,G}. Epilogue fuses tanh*sigmoid gating,
// the f_w dot (shfl-xor row reduce) and atomicAdd of partial scores.
#define TK     64
#define SA_LD2 72  // LDS row stride in bf16 (144B): conflict-free b128 reads

__global__ void __launch_bounds__(256)
box_gemm_kernel(const float* __restrict__ box, const unsigned short* __restrict__ Wpk,
                const unsigned short* __restrict__ Wppk, const float* __restrict__ qy,
                const float* __restrict__ qg, const float* __restrict__ f_w,
                float* __restrict__ S) {
  __shared__ alignas(16) unsigned short sA[2][16 * SA_LD2];

  int r0 = blockIdx.x << 4;
  int wid = threadIdx.x >> 5;
  int lane = threadIdx.x & 31;
  int m = lane & 15, half = lane >> 4;

  v8f aY[4] = {};
  v8f aG[4] = {};

  // staging coords: 16 rows x 64 K per tile; 4 consecutive floats per thread
  int srow = threadIdx.x >> 4;          // 0..15
  int scol = (threadIdx.x & 15) << 2;   // 0..60 step 4
  const float* srcp = box + (long)(r0 + srow) * I_CNN + scol;

  float4 stg = *(const float4*)srcp;  // k0 = 0
  int p = 0;

#pragma unroll 1
  for (int k0 = 0; k0 < I_CNN; k0 += TK) {
    // commit staged tile to LDS buffer p
    v4us o;
    o[0] = f2bf(stg.x); o[1] = f2bf(stg.y); o[2] = f2bf(stg.z); o[3] = f2bf(stg.w);
    *(v4us*)&sA[p][srow * SA_LD2 + scol] = o;
    __syncthreads();

    // prefetch next global stage under this tile's WMMAs
    if (k0 + TK < I_CNN) stg = *(const float4*)(srcp + k0 + TK);

#pragma unroll
    for (int kk = 0; kk < TK; kk += 32) {
      const unsigned short* ap = &sA[p][m * SA_LD2 + kk + half * 8];
      v16bf a = frag_from(ap, ap + 16);
      v16bf by[4], bg[4];
#pragma unroll
      for (int i = 0; i < 4; ++i) {
        int n0 = ((wid << 2) + i) << 4;
        by[i] = load_b_bf(Wpk,  LDW, n0, k0 + kk);
        bg[i] = load_b_bf(Wppk, LDW, n0, k0 + kk);
      }
#pragma unroll
      for (int i = 0; i < 4; ++i) {
        aY[i] = wmma_bf16(a, by[i], aY[i]);
        aG[i] = wmma_bf16(a, bg[i], aG[i]);
      }
    }
    p ^= 1;
  }

  int bb = r0 >> 7;  // 16-row tile lies within one batch entry (128 boxes)
#pragma unroll
  for (int i = 0; i < 4; ++i) {
    int n0 = ((wid << 2) + i) << 4;
    float fw  = f_w[n0 + m];
    float qyv = qy[(long)bb * HID + n0 + m];
    float qgv = qg[(long)bb * HID + n0 + m];
#pragma unroll
    for (int r = 0; r < 8; ++r) {
      float si = tanhf(aY[i][r] + qyv) * sigmoidf_(aG[i][r] + qgv);
      float pp = si * fw;
      pp += __shfl_xor(pp, 1, 16);
      pp += __shfl_xor(pp, 2, 16);
      pp += __shfl_xor(pp, 4, 16);
      pp += __shfl_xor(pp, 8, 16);
      if (m == 0) atomicAdd(&S[r0 + r + 8 * half], pp);
    }
  }
}

// counts[b] = sum_{m < index[b]} sigmoid(S[b,m])
__global__ void __launch_bounds__(256)
finalize_kernel(const float* __restrict__ S, const long long* __restrict__ index,
                float* __restrict__ out) {
  int b = blockIdx.x * blockDim.x + threadIdx.x;
  if (b >= BB) return;
  long long lim = index[b];  // reference dtype is int64
  float acc = 0.0f;
#pragma unroll 4
  for (int m = 0; m < MM; ++m) {
    float sc = sigmoidf_(S[b * MM + m]);
    if ((long long)m < lim) acc += sc;
  }
  out[b] = acc;
}

// ---------------------------------------------------------------- launcher

extern "C" void kernel_launch(void* const* d_in, const int* in_sizes, int n_in,
                              void* d_out, int out_size, void* d_ws, size_t ws_size,
                              hipStream_t stream) {
  (void)in_sizes; (void)n_in; (void)out_size; (void)ws_size;
  // setup_inputs order: 0 wholefeat, 1 pooled, 2 box_feats, 3 q_feats,
  // 4 box_coords, 5 index, 6 w_ih, 7 w_hh, 8 b_ih, 9 b_hh,
  // 10 W_w, 11 W_b, 12 Wp_w, 13 Wp_b, 14 f_w
  const float*     box_feats = (const float*)d_in[2];
  const float*     q_feats   = (const float*)d_in[3];
  const long long* index     = (const long long*)d_in[5];
  const float*     w_ih      = (const float*)d_in[6];
  const float*     w_hh      = (const float*)d_in[7];
  const float*     b_ih      = (const float*)d_in[8];
  const float*     b_hh      = (const float*)d_in[9];
  const float*     W_w       = (const float*)d_in[10];
  const float*     W_b       = (const float*)d_in[11];
  const float*     Wp_w      = (const float*)d_in[12];
  const float*     Wp_b      = (const float*)d_in[13];
  const float*     f_w       = (const float*)d_in[14];
  float* out = (float*)d_out;

  // Workspace carve (256B aligned). Zeroed prefix: h0(bf16), c(f32), S(f32).
  char* w = (char*)d_ws;
  auto carve = [&](size_t bytes) {
    char* p = w;
    w += (bytes + 255) & ~(size_t)255;
    return p;
  };
  unsigned short* h0    = (unsigned short*)carve((size_t)BB * HL * 2);
  float*          cbuf  = (float*)carve((size_t)BB * HL * 4);
  float*          S     = (float*)carve((size_t)BB * MM * 4);
  size_t zero_bytes = (size_t)(w - (char*)d_ws);
  unsigned short* h1    = (unsigned short*)carve((size_t)BB * HL * 2);
  float*          qy    = (float*)carve((size_t)BB * HID * 4);
  float*          qg    = (float*)carve((size_t)BB * HID * 4);
  unsigned short* wihpk = (unsigned short*)carve((size_t)4 * HL * QPAD * 2);
  unsigned short* whhpk = (unsigned short*)carve((size_t)4 * HL * HL * 2);
  unsigned short* Wpk   = (unsigned short*)carve((size_t)HID * LDW * 2);
  unsigned short* Wppk  = (unsigned short*)carve((size_t)HID * LDW * 2);
  unsigned short* xpad  = (unsigned short*)carve((size_t)TT * BB * QPAD * 2);

  hipMemsetAsync(d_ws, 0, zero_bytes, stream);

  // One-time bf16 repacks (inner GEMM loops then run conversion-free).
  {
    long n_whh = (long)4 * HL * HL;
    pack_bf16_kernel<<<(int)(n_whh / 1024), 256, 0, stream>>>(w_hh, whhpk, n_whh);
    long n_W = (long)HID * LDW;
    pack_bf16_kernel<<<(int)(n_W / 1024), 256, 0, stream>>>(W_w, Wpk, n_W);
    pack_bf16_kernel<<<(int)(n_W / 1024), 256, 0, stream>>>(Wp_w, Wppk, n_W);
    long n_wih = (long)4 * HL * QPAD;
    pack_pad_kernel<<<(int)((n_wih + 255) / 256), 256, 0, stream>>>(
        w_ih, wihpk, 4 * HL, QEMB, QPAD);
    long n_x = (long)TT * BB * QPAD;
    pack_x_kernel<<<(int)((n_x + 255) / 256), 256, 0, stream>>>(q_feats, xpad);
  }

  // LSTM: 20 stream-ordered steps, ping-ponged bf16 hidden state.
  unsigned short* hb[2] = {h0, h1};
  for (int t = 0; t < TT; ++t) {
    // 1024 waves: 16 batch tiles x 64 hidden tiles; 4 waves/block
    lstm_step_kernel<<<256, 128, 0, stream>>>(xpad, wihpk, whhpk, b_ih, b_hh,
                                              hb[t & 1], hb[(t + 1) & 1], cbuf, t);
  }
  const unsigned short* hfin = hb[TT & 1];  // t=19 wrote hb[0]

  // 512 waves: 16 batch tiles x 32 col tiles
  qproj_kernel<<<128, 128, 0, stream>>>(hfin, Wpk, W_b, Wppk, Wp_b, qy, qg);

  // One block per 16-row tile (2048 blocks), all 512 columns per block.
  box_gemm_kernel<<<2048, 256, 0, stream>>>(box_feats, Wpk, Wppk, qy, qg, f_w, S);

  finalize_kernel<<<1, 256, 0, stream>>>(S, index, out);
}
